// SAModule_32169305047370
// MI455X (gfx1250) — compile-verified, matrix operand order checked by reading, not run on hardware
//
#include <hip/hip_runtime.h>

// ---------------- problem constants (match reference) ----------------
#define BC    8            // clouds
#define NP    2048         // points per cloud
#define SS    1024         // fps samples per cloud
#define KK    64           // max neighbors
#define R2C   0.04f        // radius^2
#define CIN   64
#define H1C   128
#define COUT  256
#define CK1   96           // 67 padded up to 3 x 32 for WMMA k-steps
#define BSS   (BC * SS)    // 8192 centers

typedef __attribute__((ext_vector_type(16))) _Float16 v16h;
typedef __attribute__((ext_vector_type(8)))  _Float16 v8h;
typedef __attribute__((ext_vector_type(8)))  float    v8f;

// -------------------------------------------------------------------
// Kernel 1: farthest point sampling, one workgroup per cloud.
// -------------------------------------------------------------------
__global__ __launch_bounds__(256) void fps_kernel(const float* __restrict__ pos,
                                                  int* __restrict__ idx) {
  const int b = blockIdx.x;
  const float* P = pos + (size_t)b * NP * 3;
  __shared__ float dist[NP];
  __shared__ float redv[256];
  __shared__ int   redi[256];
  __shared__ float curp[3];
  const int t = threadIdx.x;

  const float p0x = P[0], p0y = P[1], p0z = P[2];
  for (int i = t; i < NP; i += 256) {
    float dx = P[i * 3 + 0] - p0x;
    float dy = P[i * 3 + 1] - p0y;
    float dz = P[i * 3 + 2] - p0z;
    dist[i] = dx * dx + dy * dy + dz * dz;
  }
  if (t == 0) idx[b * SS] = 0;
  __syncthreads();

  for (int s = 1; s < SS; ++s) {
    float bm = -1.0f; int bi = 0;
    for (int i = t; i < NP; i += 256) {
      float d = dist[i];
      if (d > bm) { bm = d; bi = i; }
    }
    redv[t] = bm; redi[t] = bi;
    __syncthreads();
    for (int off = 128; off > 0; off >>= 1) {
      if (t < off && redv[t + off] > redv[t]) {
        redv[t] = redv[t + off]; redi[t] = redi[t + off];
      }
      __syncthreads();
    }
    if (t == 0) {
      int nxt = redi[0];
      idx[b * SS + s] = nxt;
      curp[0] = P[nxt * 3 + 0];
      curp[1] = P[nxt * 3 + 1];
      curp[2] = P[nxt * 3 + 2];
    }
    __syncthreads();
    const float cx = curp[0], cy = curp[1], cz = curp[2];
    for (int i = t; i < NP; i += 256) {
      float dx = P[i * 3 + 0] - cx;
      float dy = P[i * 3 + 1] - cy;
      float dz = P[i * 3 + 2] - cz;
      float d = dx * dx + dy * dy + dz * dz;
      if (d < dist[i]) dist[i] = d;
    }
    __syncthreads();
  }
}

// -------------------------------------------------------------------
// Kernel 2: radius ball query, one wave32 per center. Compaction via
// per-wave LDS atomic counter (ds_add_rtn_u32). Also emits q / batch.
// -------------------------------------------------------------------
__global__ __launch_bounds__(256) void ballq_kernel(const float* __restrict__ pos,
                                                    const int* __restrict__ idx,
                                                    int* __restrict__ nbr,
                                                    int* __restrict__ cnt,
                                                    float* __restrict__ outq,
                                                    int* __restrict__ outb) {
  __shared__ int wcount[8];
  const int lane = threadIdx.x & 31;
  const int wave = threadIdx.x >> 5;
  if (lane == 0) wcount[wave] = 0;
  __syncthreads();

  const int wid = (blockIdx.x * blockDim.x + threadIdx.x) >> 5;   // center id
  const int b = wid / SS;
  const float* P = pos + (size_t)b * NP * 3;
  const int qi = idx[wid];
  const float qx = P[qi * 3 + 0];
  const float qy = P[qi * 3 + 1];
  const float qz = P[qi * 3 + 2];
  if (lane < 3) outq[wid * 3 + lane] = (lane == 0) ? qx : ((lane == 1) ? qy : qz);
  if (lane == 0) outb[wid] = b;

  int* nb = nbr + (size_t)wid * KK;
  for (int base = 0; base < NP; base += 32) {
    const int p = base + lane;
    float dx = P[p * 3 + 0] - qx;
    float dy = P[p * 3 + 1] - qy;
    float dz = P[p * 3 + 2] - qz;
    float d2 = dx * dx + dy * dy + dz * dz;
    if (d2 <= R2C) {
      int slot = atomicAdd(&wcount[wave], 1);
      if (slot < KK) nb[slot] = p;
    }
  }
  int total = atomicAdd(&wcount[wave], 0);    // read current value
  if (lane == 0) cnt[wid] = (total < KK) ? total : KK;
}

// -------------------------------------------------------------------
// Kernel 3: weight transpose + f16 conversion into workspace.
// w1t: [H1][CK1]  (N-major, k padded 67->96), w2t: [COUT][H1].
// -------------------------------------------------------------------
__global__ __launch_bounds__(256) void wconv_kernel(const float* __restrict__ w1,
                                                    const float* __restrict__ w2,
                                                    _Float16* __restrict__ w1t,
                                                    _Float16* __restrict__ w2t) {
  const int i = blockIdx.x * 256 + threadIdx.x;
  if (i < H1C * CK1) {
    int n = i / CK1, k = i % CK1;
    w1t[i] = (_Float16)((k < CIN + 3) ? w1[k * H1C + n] : 0.0f);
  }
  if (i < COUT * H1C) {
    int n = i / H1C, k = i % H1C;
    w2t[i] = (_Float16)w2[k * COUT + n];
  }
}

// -------------------------------------------------------------------
// WMMA fragment helpers. 16-bit A-matrix 16x32 striping (ISA 7.12.2):
// lane<16 -> K pairs {0..7,16..23}, lane>=16 -> {8..15,24..31}, i.e.
// two contiguous 16B chunks per lane per tile. N-major weight storage
// makes B fragments load with the identical pattern.
// -------------------------------------------------------------------
static __device__ __forceinline__ v16h load_frag(const _Float16* base, int stride,
                                                 int r0, int k0, int lane) {
  const int row = r0 + (lane & 15);
  const int grp = (lane >> 4) << 3;          // 0 or 8 halfs
  const _Float16* p = base + (size_t)row * stride + k0 + grp;
  union { v16h v; v8h h[2]; } u;
  u.h[0] = *(const v8h*)(p);
  u.h[1] = *(const v8h*)(p + 16);
  return u.v;
}

static __device__ __forceinline__ v8f wmma_f16(v16h a, v16h b, v8f c) {
  return __builtin_amdgcn_wmma_f32_16x16x32_f16(false, a, false, b,
                                                (short)0, c, false, false);
}

// -------------------------------------------------------------------
// Kernel 4: fused gather -> MLP1(WMMA) -> ReLU -> MLP2(WMMA) -> masked
// row-max. One workgroup (8 waves) per center; M=64 edge rows.
// -------------------------------------------------------------------
__global__ __launch_bounds__(256) void mlpmax_kernel(const float* __restrict__ x,
                                                     const float* __restrict__ pos,
                                                     const int* __restrict__ idx,
                                                     const int* __restrict__ nbr,
                                                     const int* __restrict__ cnt,
                                                     const _Float16* __restrict__ w1t,
                                                     const _Float16* __restrict__ w2t,
                                                     const float* __restrict__ b1,
                                                     const float* __restrict__ b2,
                                                     float* __restrict__ out) {
  __shared__ __align__(16) _Float16 featA[64 * CK1];   // 12 KB
  __shared__ __align__(16) _Float16 hA[64 * H1C];      // 16 KB

  const int c = blockIdx.x;                 // center id
  const int b = c / SS;
  const int t = threadIdx.x;
  const int wave = t >> 5;
  const int lane = t & 31;

  __builtin_prefetch(w1t, 0, 1);            // -> global_prefetch_b8
  __builtin_prefetch(w2t, 0, 1);

  const int myc = cnt[c];
  const int qi = idx[c];
  const float* P = pos + (size_t)b * NP * 3;
  const float qx = P[qi * 3 + 0];
  const float qy = P[qi * 3 + 1];
  const float qz = P[qi * 3 + 2];

  // ---- gather edge features: 4 threads per row, 24 cols each ----
  {
    const int r = t >> 2;
    const int q4 = t & 3;
    const int nv = (r < myc) ? nbr[(size_t)c * KK + r] : -1;
    for (int col = q4 * 24; col < q4 * 24 + 24; ++col) {
      float v = 0.0f;
      if (nv >= 0) {
        if (col < CIN)       v = x[((size_t)b * NP + nv) * CIN + col];
        else if (col == 64)  v = P[nv * 3 + 0] - qx;
        else if (col == 65)  v = P[nv * 3 + 1] - qy;
        else if (col == 66)  v = P[nv * 3 + 2] - qz;
      }
      featA[r * CK1 + col] = (_Float16)v;
    }
  }
  __syncthreads();

  // ---- GEMM1: [64 x 96] @ [96 x 128] -> ReLU -> hA (f16, LDS) ----
  {
    const int n0 = wave * 16;
    const int ncol = n0 + (lane & 15);
    const float bias = b1[ncol];
#pragma unroll
    for (int mt = 0; mt < 4; ++mt) {
      v8f acc = {};
#pragma unroll
      for (int kt = 0; kt < 3; ++kt) {
        v16h a  = load_frag(featA, CK1, mt * 16, kt * 32, lane);
        v16h bf = load_frag(w1t,  CK1, n0,      kt * 32, lane);
        acc = wmma_f16(a, bf, acc);
      }
      const int mbase = mt * 16 + ((lane >> 4) << 3);
#pragma unroll
      for (int rr = 0; rr < 8; ++rr) {
        float v = acc[rr] + bias;
        v = (v > 0.0f) ? v : 0.0f;
        hA[(mbase + rr) * H1C + ncol] = (_Float16)v;
      }
    }
  }
  __syncthreads();

  // ---- GEMM2: [64 x 128] @ [128 x 256] fused with masked row-max ----
#pragma unroll
  for (int nt = 0; nt < 2; ++nt) {
    const int n0 = (wave * 2 + nt) * 16;
    const int ncol = n0 + (lane & 15);
    float rmax = -3.0e38f;
#pragma unroll
    for (int mt = 0; mt < 4; ++mt) {
      v8f acc = {};
#pragma unroll
      for (int kt = 0; kt < 4; ++kt) {
        v16h a  = load_frag(hA,  H1C, mt * 16, kt * 32, lane);
        v16h bf = load_frag(w2t, H1C, n0,      kt * 32, lane);
        acc = wmma_f16(a, bf, acc);
      }
      const int mbase = mt * 16 + ((lane >> 4) << 3);
#pragma unroll
      for (int rr = 0; rr < 8; ++rr) {
        if ((mbase + rr) < myc && acc[rr] > rmax) rmax = acc[rr];
      }
    }
    // combine lane l with lane l^16 (SWAPX16 via ds_swizzle, wave32)
    {
      int o = __builtin_amdgcn_ds_swizzle(__float_as_int(rmax), 0x401F);
      float other = __int_as_float(o);
      if (other > rmax) rmax = other;
    }
    if (lane < 16) {
      float v = (rmax <= -3.0e38f) ? 0.0f : (rmax + b2[ncol]);
      out[(size_t)c * COUT + ncol] = v;
    }
  }
}

// -------------------------------------------------------------------
// Host launcher.
//   inputs:  x, pos, batch, w1, b1, w2, b2
//   d_out:   out[8192*256] f32 | q[8192*3] f32 | batch_out[8192] i32
//   d_ws:    idx | nbr | cnt | w1t(f16) | w2t(f16)
// -------------------------------------------------------------------
extern "C" void kernel_launch(void* const* d_in, const int* in_sizes, int n_in,
                              void* d_out, int out_size, void* d_ws, size_t ws_size,
                              hipStream_t stream) {
  const float* x   = (const float*)d_in[0];
  const float* pos = (const float*)d_in[1];
  // d_in[2] = batch (unused; batch_out is recomputed)
  const float* w1  = (const float*)d_in[3];
  const float* b1  = (const float*)d_in[4];
  const float* w2  = (const float*)d_in[5];
  const float* b2  = (const float*)d_in[6];

  float* out  = (float*)d_out;
  float* outq = out + (size_t)BSS * COUT;
  int*   outb = (int*)(out + (size_t)BSS * COUT + (size_t)BSS * 3);

  char* ws = (char*)d_ws;
  int* idx = (int*)ws;                 ws += (size_t)BSS * sizeof(int);
  int* nbr = (int*)ws;                 ws += (size_t)BSS * KK * sizeof(int);
  int* cnt = (int*)ws;                 ws += (size_t)BSS * sizeof(int);
  _Float16* w1t = (_Float16*)ws;       ws += (size_t)H1C * CK1 * sizeof(_Float16);
  _Float16* w2t = (_Float16*)ws;       ws += (size_t)COUT * H1C * sizeof(_Float16);

  fps_kernel  <<<BC,        256, 0, stream>>>(pos, idx);
  ballq_kernel<<<BSS / 8,   256, 0, stream>>>(pos, idx, nbr, cnt, outq, outb);
  wconv_kernel<<<128,       256, 0, stream>>>(w1, w2, w1t, w2t);
  mlpmax_kernel<<<BSS,      256, 0, stream>>>(x, pos, idx, nbr, cnt,
                                              w1t, w2t, b1, b2, out);
}
